// MultiHeadAttention_81870666596431
// MI455X (gfx1250) — matrix operand order repaired and checked by reading, not compiled, not run on hardware
//
#include <hip/hip_runtime.h>
#include <hip/hip_bf16.h>
#include <stdint.h>

// ---------------- problem constants ----------------
#define Bn  4
#define Sn  2048
#define Dn  2048
#define Hn  16
#define DKn 128

typedef __attribute__((ext_vector_type(16))) _Float16 v16h;
typedef __attribute__((ext_vector_type(8)))  _Float16 v8h;
typedef __attribute__((ext_vector_type(4)))  _Float16 v4h;
typedef __attribute__((ext_vector_type(8)))  float    v8f;
typedef __attribute__((ext_vector_type(4)))  float    v4f;

__device__ __forceinline__ v16h mk16(v8h lo, v8h hi) {
  union { v16h w; v8h h[2]; } u;
  u.h[0] = lo; u.h[1] = hi;
  return u.w;
}

// CDNA5 async global->LDS (ASYNCcnt) — per-lane fallback path.
__device__ __forceinline__ void async_load_b128(const void* lds_ptr, const void* gptr) {
  uint32_t lds = (uint32_t)(uintptr_t)lds_ptr;
  unsigned long long g = (unsigned long long)(uintptr_t)gptr;
  asm volatile("global_load_async_to_lds_b128 %0, %1, off"
               :: "v"(lds), "v"(g) : "memory");
}
__device__ __forceinline__ void wait_async0() {
  asm volatile("s_wait_asynccnt 0x0" ::: "memory");
}

// ---------------- Tensor Data Mover: whole-tile DMA global->LDS ----------------
#if __has_builtin(__builtin_amdgcn_tensor_load_to_lds)
#define HAVE_TDM 1
typedef unsigned int u32x4 __attribute__((ext_vector_type(4)));
typedef int          i32x4 __attribute__((ext_vector_type(4)));
typedef int          i32x8 __attribute__((ext_vector_type(8)));

// 2D tile load, data_size = 2B. pad_interval_enc: pad every 2^(enc+1) DWORDs
// stored; pad_amount_enc: add (enc+1) DWORDs to the LDS address.
__device__ __forceinline__ void tdm_load_2d(
    const void* lds_ptr, const void* gptr,
    unsigned tensor_d0, unsigned tensor_d1,
    unsigned tile_d0, unsigned tile_d1,
    unsigned long long d0_stride,
    unsigned pad_interval_enc, unsigned pad_amount_enc) {
  unsigned lds = (unsigned)(uintptr_t)lds_ptr;           // LDS offset = addr[31:0]
  unsigned long long ga = (unsigned long long)(uintptr_t)gptr;
  u32x4 g0;
  g0[0] = 1u;                                            // count=1 (valid D#)
  g0[1] = lds;                                           // lds_addr
  g0[2] = (unsigned)(ga & 0xffffffffu);                  // global_addr[31:0]
  g0[3] = (unsigned)((ga >> 32) & 0x1ffffffu) | (2u << 30);  // [120:96] + type=2
  i32x8 g1;
  g1[0] = (int)((1u << 16)                               // data_size = 2 bytes
              | (1u << 20)                               // pad_enable
              | (pad_interval_enc << 22)
              | (pad_amount_enc << 25));
  g1[1] = (int)((tensor_d0 & 0xffffu) << 16);            // tensor_dim0[15:0]
  g1[2] = (int)((tensor_d0 >> 16) | ((tensor_d1 & 0xffffu) << 16));
  g1[3] = (int)((tensor_d1 >> 16) | (tile_d0 << 16));    // tile_dim0
  g1[4] = (int)(tile_d1 & 0xffffu);                      // tile_dim1 (tile_dim2=0)
  g1[5] = (int)(d0_stride & 0xffffffffu);                // tensor_dim0_stride
  g1[6] = (int)((d0_stride >> 32) & 0xffffu);
  g1[7] = 0;
  i32x4 z4 = {0, 0, 0, 0};
#if defined(__clang_major__) && (__clang_major__ >= 23)
  i32x8 z8 = {0, 0, 0, 0, 0, 0, 0, 0};
  __builtin_amdgcn_tensor_load_to_lds(g0, g1, z4, z4, z8, 0);
#else
  __builtin_amdgcn_tensor_load_to_lds(g0, g1, z4, z4, 0);
#endif
}
__device__ __forceinline__ void wait_tensor0() {
  __builtin_amdgcn_s_wait_tensorcnt(0);
}
#endif

// ---------------- fp32 -> f16 weight conversion ----------------
__global__ void cvt_f32_f16(const float* __restrict__ src,
                            _Float16* __restrict__ dst, int n) {
  int i = (blockIdx.x * blockDim.x + threadIdx.x) * 4;
  if (i < n) {
    v4f x = *(const v4f*)(src + i);
    v4h y;
    y[0] = (_Float16)x[0]; y[1] = (_Float16)x[1];
    y[2] = (_Float16)x[2]; y[3] = (_Float16)x[3];
    *(v4h*)(dst + i) = y;
  }
}

// ---------------- WMMA GEMM: Y = X(f32)[M,K] @ W(f16)[N,K]^T + bias ----------------
// 128M x 256N block tile, BK=32, 8 waves (2Mx4N), wave tile 64x64 ->
// 16 wmma per k-step from 8 fragment loads. Double-buffered LDS; B tile via
// TDM (one DMA per k-step), A tile software-pipelined through VGPRs (cvt f32->f16).
template <bool F16OUT>
__global__ __launch_bounds__(256) void gemm_xwT(
    const float* __restrict__ X, const _Float16* __restrict__ W,
    const float* __restrict__ bias, void* __restrict__ Yv,
    int M, int N, int K) {
  const int LDA = 40, LDB = 40;
  __shared__ _Float16 As[2][128 * 40];
  __shared__ _Float16 Bs[2][256 * 40];

  int tid  = threadIdx.x;
  int lane = tid & 31, wave = tid >> 5;
  int hl = lane >> 4, ln = lane & 15;
  int m0 = blockIdx.y * 128, n0 = blockIdx.x * 256;
  int wm = wave >> 2, wn = wave & 3;     // 2 (M) x 4 (N) wave grid
  int mw = wm * 64,   nw = wn * 64;

  int acol = (tid & 7) * 4;
  int arb  = tid >> 3;

  v8f acc[4][4] = {};

  auto loadB = [&](int k0, int bb) {
#if HAVE_TDM
    if (tid == 0)
      tdm_load_2d(Bs[bb], W + (size_t)n0 * K + k0,
                  (unsigned)K, (unsigned)N, 32u, 256u,
                  (unsigned long long)K,
                  /*row = 64B = 16 DW -> enc 3*/ 3u,
                  /*pad = 16B = 4 DW -> enc 3*/ 3u);
#else
#pragma unroll
    for (int i = 0; i < 4; i++) {
      int idx = (i * 256 + tid) * 8;
      int r = idx >> 5, c = idx & 31;
      async_load_b128(Bs[bb] + r * LDB + c,
                      W + (size_t)(n0 + r) * K + k0 + c);
    }
#endif
  };
  auto waitB = [&]() {
#if HAVE_TDM
    wait_tensor0();
#else
    wait_async0();
#endif
  };
  auto loadA_regs = [&](int k0, v4f xr[4]) {
#pragma unroll
    for (int i = 0; i < 4; i++)
      xr[i] = *(const v4f*)(X + (size_t)(m0 + arb + i * 32) * K + k0 + acol);
  };
  auto storeA = [&](v4f xr[4], int bb) {
#pragma unroll
    for (int i = 0; i < 4; i++) {
      v4h y;
      y[0] = (_Float16)xr[i][0]; y[1] = (_Float16)xr[i][1];
      y[2] = (_Float16)xr[i][2]; y[3] = (_Float16)xr[i][3];
      *(v4h*)(As[bb] + (arb + i * 32) * LDA + acol) = y;
    }
  };

  // prologue: stage k-step 0 into buffer 0
  {
    v4f xr[4];
    loadA_regs(0, xr);
    loadB(0, 0);
    storeA(xr, 0);
  }

  const int KS = K / 32;
  for (int ks = 0; ks < KS; ks++) {
    int bb = ks & 1;
    waitB();
    __syncthreads();

    v4f xr[4];
    bool hasnext = (ks + 1 < KS);
    if (hasnext) {
      loadB((ks + 1) * 32, bb ^ 1);     // DMA into other buffer
      loadA_regs((ks + 1) * 32, xr);    // global loads overlap compute
    }

    // batch all fragment loads, then burst 16 WMMAs
    v16h bfrag[4], afrag[4];
#pragma unroll
    for (int nf = 0; nf < 4; nf++) {
      const _Float16* bp = Bs[bb] + (nw + nf * 16 + ln) * LDB + hl * 8;
      bfrag[nf] = mk16(*(const v8h*)bp, *(const v8h*)(bp + 16));
    }
#pragma unroll
    for (int mf = 0; mf < 4; mf++) {
      const _Float16* ap = As[bb] + (mw + mf * 16 + ln) * LDA + hl * 8;
      afrag[mf] = mk16(*(const v8h*)ap, *(const v8h*)(ap + 16));
    }
#pragma unroll
    for (int mf = 0; mf < 4; mf++)
#pragma unroll
      for (int nf = 0; nf < 4; nf++)
        acc[mf][nf] = __builtin_amdgcn_wmma_f32_16x16x32_f16(
            false, afrag[mf], false, bfrag[nf], (short)0, acc[mf][nf], false, false);

    if (hasnext) storeA(xr, bb ^ 1);    // cvt + LDS store after compute
  }

  // epilogue: + bias, store
#pragma unroll
  for (int nf = 0; nf < 4; nf++) {
    int col = n0 + nw + nf * 16 + ln;
    float bv = bias[col];
#pragma unroll
    for (int mf = 0; mf < 4; mf++) {
#pragma unroll
      for (int r = 0; r < 8; r++) {
        int row = m0 + mw + mf * 16 + r + hl * 8;
        float val = acc[mf][nf][r] + bv;
        if (F16OUT)
          ((_Float16*)Yv)[(size_t)row * N + col] = (_Float16)val;
        else
          ((float*)Yv)[(size_t)row * N + col] = val;
      }
    }
  }
}

// ---------------- flash attention over f16 q/k/v, f32 ctx out ----------------
// grid: (S/64, H, B); block: 128 threads = 4 waves; each wave owns 16 query rows.
__global__ __launch_bounds__(128) void flash_attn(
    const _Float16* __restrict__ qg, const _Float16* __restrict__ kg,
    const _Float16* __restrict__ vg, const int* __restrict__ maskg,
    float* __restrict__ ctx) {
  const int LKS = 136;  // K-tile row stride (halves): 128 + 8 pad
  const int LVT = 72;   // Vt row stride (halves)
  __shared__ _Float16 Ks[64 * 136];
  __shared__ _Float16 Vt[128 * 72];
  __shared__ float    mb[64];
  __shared__ _Float16 Ps[4][16 * 64];

  int tid  = threadIdx.x;
  int lane = tid & 31, wave = tid >> 5;
  int hl = lane >> 4, ln = lane & 15;
  int b = blockIdx.z, h = blockIdx.y;
  int q0 = blockIdx.x * 64 + wave * 16;
  size_t headoff = (size_t)h * DKn;

  // Q fragments for this wave's 16 rows: 16 x 128 (4 K-fragments)
  v16h qf[4];
  {
    const _Float16* qbase = qg + ((size_t)(b * Sn) + q0 + ln) * Dn + headoff;
#pragma unroll
    for (int kf = 0; kf < 4; kf++) {
      const _Float16* p = qbase + kf * 32 + hl * 8;
      qf[kf] = mk16(*(const v8h*)p, *(const v8h*)(p + 16));
    }
  }

  v8f Of[8] = {};
  float mrow[8], lrow[8];
#pragma unroll
  for (int r = 0; r < 8; r++) { mrow[r] = -__builtin_inff(); lrow[r] = 0.f; }

  for (int kt = 0; kt < Sn; kt += 64) {
    __syncthreads();
    if (tid < 64)
      mb[tid] = maskg[b * Sn + kt + tid] ? 0.f : -__builtin_inff();
    // K tile 64x128 row-major: one TDM DMA (or per-lane async fallback)
#if HAVE_TDM
    if (tid == 0)
      tdm_load_2d(Ks, kg + ((size_t)(b * Sn) + kt) * Dn + headoff,
                  (unsigned)Dn, (unsigned)Sn, 128u, 64u,
                  (unsigned long long)Dn,
                  /*row = 256B = 64 DW -> enc 5*/ 5u,
                  /*pad = 16B  =  4 DW -> enc 3*/ 3u);
#else
#pragma unroll
    for (int i = 0; i < 4; i++) {
      int idx = (i * 128 + tid) * 16;
      int r = idx >> 7, c = idx & 127;
      const _Float16* src = kg + ((size_t)(b * Sn) + kt + r) * Dn + headoff + c;
      _Float16* dst = Ks + r * LKS + c;
      async_load_b128(dst,     src);
      async_load_b128(dst + 8, src + 8);
    }
#endif
    // V tile transposed: Vt[dk][key] (needs scatter -> VGPR path)
#pragma unroll
    for (int i = 0; i < 4; i++) {
      int idx = (i * 128 + tid) * 16;
      int r = idx >> 7, c = idx & 127;
      const _Float16* src = vg + ((size_t)(b * Sn) + kt + r) * Dn + headoff + c;
      v8h x0 = *(const v8h*)src;
      v8h x1 = *(const v8h*)(src + 8);
#pragma unroll
      for (int e = 0; e < 8; e++) {
        Vt[(c + e) * LVT + r]     = x0[e];
        Vt[(c + 8 + e) * LVT + r] = x1[e];
      }
    }
#if HAVE_TDM
    wait_tensor0();
#else
    wait_async0();
#endif
    __syncthreads();

    // scores S = q @ k^T (16x64): ping-pong fragment pipeline
    auto loadSB = [&](int nf, v16h bf[4]) {
#pragma unroll
      for (int kf = 0; kf < 4; kf++) {
        const _Float16* bp = Ks + (nf * 16 + ln) * LKS + kf * 32 + hl * 8;
        bf[kf] = mk16(*(const v8h*)bp, *(const v8h*)(bp + 16));
      }
    };
    v8f sf[4];
    v16h bfA[4], bfB[4];
    loadSB(0, bfA);
#pragma unroll
    for (int nf = 0; nf < 4; nf++) {
      v16h* cur = (nf & 1) ? bfB : bfA;
      v16h* nxt = (nf & 1) ? bfA : bfB;
      if (nf < 3) loadSB(nf + 1, nxt);   // prefetch next N-frag's B data
      v8f acc = {};
#pragma unroll
      for (int kf = 0; kf < 4; kf++)
        acc = __builtin_amdgcn_wmma_f32_16x16x32_f16(
            false, qf[kf], false, cur[kf], (short)0, acc, false, false);
      float biasv = mb[nf * 16 + ln];
#pragma unroll
      for (int r = 0; r < 8; r++) acc[r] += biasv;
      sf[nf] = acc;
    }

    // online softmax (per C-layout row; reductions over 16-lane halves)
    float corr[8];
#pragma unroll
    for (int r = 0; r < 8; r++) {
      float mx = fmaxf(fmaxf(sf[0][r], sf[1][r]), fmaxf(sf[2][r], sf[3][r]));
#pragma unroll
      for (int off = 1; off < 16; off <<= 1)
        mx = fmaxf(mx, __shfl_xor(mx, off, 32));
      float mnew = fmaxf(mrow[r], mx);
      float mcl  = fmaxf(mnew,    -1.0e30f);
      float mo   = fmaxf(mrow[r], -1.0e30f);
      corr[r] = __expf(mo - mcl);
      mrow[r] = mnew;
      float rs = 0.f;
#pragma unroll
      for (int nf = 0; nf < 4; nf++) {
        float p = __expf(sf[nf][r] - mcl);
        sf[nf][r] = p;
        rs += p;
      }
#pragma unroll
      for (int off = 1; off < 16; off <<= 1) rs += __shfl_xor(rs, off, 32);
      lrow[r] = lrow[r] * corr[r] + rs;
    }

    // rescale O accumulators
#pragma unroll
    for (int nf2 = 0; nf2 < 8; nf2++)
#pragma unroll
      for (int r = 0; r < 8; r++) Of[nf2][r] *= corr[r];

    // bounce P through per-wave LDS: C-layout -> A-layout, cvt to f16
    _Float16* pw = &Ps[wave][0];
#pragma unroll
    for (int nf = 0; nf < 4; nf++) {
      int col = nf * 16 + ln;
#pragma unroll
      for (int r = 0; r < 8; r++)
        pw[(r + hl * 8) * 64 + col] = (_Float16)sf[nf][r];
    }
    asm volatile("s_wait_dscnt 0x0" ::: "memory");  // same-wave LDS RAW

    v16h pf[2];
#pragma unroll
    for (int kf = 0; kf < 2; kf++) {
      const _Float16* p = pw + ln * 64 + kf * 32 + hl * 8;
      pf[kf] = mk16(*(const v8h*)p, *(const v8h*)(p + 16));
    }
    // O += P @ V: pairs of N-frags, ping-pong pipelined B fragments
    auto loadVB = [&](int np, v16h bf[4]) {
#pragma unroll
      for (int j = 0; j < 2; j++)
#pragma unroll
        for (int kf = 0; kf < 2; kf++) {
          const _Float16* bp =
              Vt + ((np * 2 + j) * 16 + ln) * LVT + kf * 32 + hl * 8;
          bf[j * 2 + kf] = mk16(*(const v8h*)bp, *(const v8h*)(bp + 16));
        }
    };
    v16h vA[4], vB[4];
    loadVB(0, vA);
#pragma unroll
    for (int np = 0; np < 4; np++) {
      v16h* cur = (np & 1) ? vB : vA;
      v16h* nxt = (np & 1) ? vA : vB;
      if (np < 3) loadVB(np + 1, nxt);
#pragma unroll
      for (int j = 0; j < 2; j++)
#pragma unroll
        for (int kf = 0; kf < 2; kf++)
          Of[np * 2 + j] = __builtin_amdgcn_wmma_f32_16x16x32_f16(
              false, pf[kf], false, cur[j * 2 + kf], (short)0, Of[np * 2 + j],
              false, false);
    }
  }

  // finalize: divide by row sums, write f32 ctx
#pragma unroll
  for (int r = 0; r < 8; r++) {
    float inv = (lrow[r] > 0.f) ? 1.f / lrow[r] : 0.f;
    int row = q0 + r + hl * 8;
#pragma unroll
    for (int nf2 = 0; nf2 < 8; nf2++) {
      int col = (int)headoff + nf2 * 16 + ln;
      ctx[((size_t)(b * Sn) + row) * Dn + col] = Of[nf2][r] * inv;
    }
  }
}

// ---------------- launch ----------------
extern "C" void kernel_launch(void* const* d_in, const int* in_sizes, int n_in,
                              void* d_out, int out_size, void* d_ws, size_t ws_size,
                              hipStream_t stream) {
  const float* Q    = (const float*)d_in[0];
  const float* K    = (const float*)d_in[1];
  const float* V    = (const float*)d_in[2];
  const int*   mask = (const int*)  d_in[3];
  const float* Wq   = (const float*)d_in[4];
  const float* bq   = (const float*)d_in[5];
  const float* Wk   = (const float*)d_in[6];
  const float* bk   = (const float*)d_in[7];
  const float* Wv   = (const float*)d_in[8];
  const float* bv   = (const float*)d_in[9];
  const float* Wo   = (const float*)d_in[10];
  const float* bo   = (const float*)d_in[11];
  float* out = (float*)d_out;

  const size_t DD  = (size_t)Dn * Dn;       // 4M
  const size_t BSD = (size_t)Bn * Sn * Dn;  // 16M

  char* ws = (char*)d_ws;
  _Float16* Wqh = (_Float16*)ws; ws += DD * 2;
  _Float16* Wkh = (_Float16*)ws; ws += DD * 2;
  _Float16* Wvh = (_Float16*)ws; ws += DD * 2;
  _Float16* Woh = (_Float16*)ws; ws += DD * 2;
  _Float16* qh  = (_Float16*)ws; ws += BSD * 2;
  _Float16* kh  = (_Float16*)ws; ws += BSD * 2;
  _Float16* vh  = (_Float16*)ws; ws += BSD * 2;
  float*    ctx = (float*)ws;    ws += BSD * 4;

  // 1) weights fp32 -> f16
  {
    int n = (int)DD;
    int blocks = (n / 4 + 255) / 256;
    cvt_f32_f16<<<blocks, 256, 0, stream>>>(Wq, Wqh, n);
    cvt_f32_f16<<<blocks, 256, 0, stream>>>(Wk, Wkh, n);
    cvt_f32_f16<<<blocks, 256, 0, stream>>>(Wv, Wvh, n);
    cvt_f32_f16<<<blocks, 256, 0, stream>>>(Wo, Woh, n);
  }

  // 2) QKV projections (f16 output)
  {
    dim3 grid(Dn / 256, (Bn * Sn) / 128);
    gemm_xwT<true><<<grid, 256, 0, stream>>>(Q, Wqh, bq, qh, Bn * Sn, Dn, Dn);
    gemm_xwT<true><<<grid, 256, 0, stream>>>(K, Wkh, bk, kh, Bn * Sn, Dn, Dn);
    gemm_xwT<true><<<grid, 256, 0, stream>>>(V, Wvh, bv, vh, Bn * Sn, Dn, Dn);
  }

  // 3) attention -> f32 ctx
  {
    dim3 grid(Sn / 64, Hn, Bn);
    flash_attn<<<grid, 128, 0, stream>>>(qh, kh, vh, mask, ctx);
  }

  // 4) output projection (f32 output)
  {
    dim3 grid(Dn / 256, (Bn * Sn) / 128);
    gemm_xwT<false><<<grid, 256, 0, stream>>>(ctx, Woh, bo, out, Bn * Sn, Dn, Dn);
  }
}
